// PromptDiversityLoss_35235911696702
// MI455X (gfx1250) — matrix operand order repaired and checked by reading, not compiled, not run on hardware
//
#include <hip/hip_runtime.h>
#include <hip/hip_bf16.h>

typedef __attribute__((ext_vector_type(16))) _Float16 v16h;
typedef __attribute__((ext_vector_type(8)))  float    v8f;
typedef __attribute__((ext_vector_type(4)))  unsigned int v4u;

#define KDIM    2048
#define NROWS   4096
#define MARGINF 0.3f

// wave block: 32 rows (2 tiles) x 64 cols (4 tiles)
#define NBI     128          // 4096 / 32 row-blocks
#define NBJ     64           // 4096 / 64 col-blocks
#define NBLOCKS 4160         // sum_{bi<128} (64 - (bi>>1))

#define TILE_STRIDE (16 * KDIM)   // halfs between consecutive 16-row tiles

union FragU { v16h h; v4u u[2]; };
union Pack8 { _Float16 h[8]; v4u u; };

// ---------------------------------------------------------------------------
// Kernel 1: L2-normalize each row, write f16 copy (row-major) to workspace.
// One 256-thread block per row. All-b128 traffic: 2x float4 in, 1x 16B out.
// ---------------------------------------------------------------------------
__global__ void pd_normalize(const float* __restrict__ x, _Float16* __restrict__ y) {
    const int row = blockIdx.x;
    const int t   = threadIdx.x;                     // 0..255, 8 cols each
    const float4* __restrict__ xr = (const float4*)(x + (size_t)row * KDIM);
    const float4 u0 = xr[2 * t];
    const float4 u1 = xr[2 * t + 1];
    float ss = u0.x * u0.x + u0.y * u0.y + u0.z * u0.z + u0.w * u0.w
             + u1.x * u1.x + u1.y * u1.y + u1.z * u1.z + u1.w * u1.w;

    __shared__ float red[256];
    red[t] = ss;
    __syncthreads();
    for (int s = 128; s > 0; s >>= 1) {
        if (t < s) red[t] += red[t + s];
        __syncthreads();
    }
    const float scale = 1.0f / fmaxf(sqrtf(red[0]), 1e-12f);

    Pack8 p;
    p.h[0] = (_Float16)(u0.x * scale);
    p.h[1] = (_Float16)(u0.y * scale);
    p.h[2] = (_Float16)(u0.z * scale);
    p.h[3] = (_Float16)(u0.w * scale);
    p.h[4] = (_Float16)(u1.x * scale);
    p.h[5] = (_Float16)(u1.y * scale);
    p.h[6] = (_Float16)(u1.z * scale);
    p.h[7] = (_Float16)(u1.w * scale);
    *(v4u*)(y + (size_t)row * KDIM + 8 * t) = p.u;
}

// ---------------------------------------------------------------------------
// Kernel 2: zero accumulators.
// ---------------------------------------------------------------------------
__global__ void pd_init(float* accf, unsigned int* accu) {
    *accf = 0.0f;
    *accu = 0u;
}

// ---------------------------------------------------------------------------
// Kernel 3: one wave per 32x64 Gram block. Register-blocked 2x4 tiles.
// Only TWO per-lane base pointers; all tile strides are compile-time
// constants folded into the 24-bit instruction offsets (no pointer spills).
// ---------------------------------------------------------------------------
__global__ void pd_blocks(const _Float16* __restrict__ P,
                          float* __restrict__ accf,
                          unsigned int* __restrict__ accu) {
    const int wid  = threadIdx.x >> 5;
    const int lane = threadIdx.x & 31;
    const int blk  = blockIdx.x * 8 + wid;           // uniform per wave
    if (blk >= NBLOCKS) return;                      // wave-uniform branch

    // map linear block id -> (bi, bj); kept blocks satisfy bj >= bi>>1
    int bi = 0, rem = blk;
    for (;;) {
        const int cnt = NBJ - (bi >> 1);
        if (rem < cnt) break;
        rem -= cnt;
        ++bi;
    }
    const int bj = (bi >> 1) + rem;

    const int i0 = bi * 32;
    const int j0 = bj * 64;

    const int m  = lane & 15;                        // row/col within tile
    const int hi = lane >> 4;                        // lane-half selector

    // A fragment (16x32 f16): K = hi*8 + [0,8) and 16 + hi*8 + [0,8)
    // B fragment (32x16 f16): K = hi*8 + [0,8) and 16 + hi*8 + [0,8)
    const _Float16* __restrict__ pa = P + (size_t)(i0 + m) * KDIM + hi * 8;
    const _Float16* __restrict__ pb = P + (size_t)(j0 + m) * KDIM + hi * 8;

    v8f acc[2][4];
#pragma unroll
    for (int rt = 0; rt < 2; ++rt)
#pragma unroll
        for (int st = 0; st < 4; ++st)
            acc[rt][st] = (v8f){};

    for (int k0 = 0; k0 < KDIM; k0 += 32) {
        FragU A[2], B[4];
#pragma unroll
        for (int rt = 0; rt < 2; ++rt) {
            A[rt].u[0] = *(const v4u*)(pa + rt * TILE_STRIDE + k0);
            A[rt].u[1] = *(const v4u*)(pa + rt * TILE_STRIDE + k0 + 16);
        }
#pragma unroll
        for (int st = 0; st < 4; ++st) {
            B[st].u[0] = *(const v4u*)(pb + st * TILE_STRIDE + k0);
            B[st].u[1] = *(const v4u*)(pb + st * TILE_STRIDE + k0 + 16);
        }
        __builtin_prefetch(pa + k0 + 64, 0, 3);
        __builtin_prefetch(pb + k0 + 64, 0, 3);

#pragma unroll
        for (int rt = 0; rt < 2; ++rt)
#pragma unroll
            for (int st = 0; st < 4; ++st)
                acc[rt][st] = __builtin_amdgcn_wmma_f32_16x16x32_f16(
                    /*neg_a=*/false, A[rt].h, /*neg_b=*/false, B[st].h,
                    /*c_mod=*/(short)0, acc[rt][st],
                    /*reuse_a=*/false, /*reuse_b=*/false);
    }

    // C layout: VGPR r holds M = hi*8 + r, N = lane&15 (per 16x16 tile)
    float        lsum = 0.0f;
    unsigned int lcnt = 0u;
#pragma unroll
    for (int rt = 0; rt < 2; ++rt) {
        const int Ibase = i0 + rt * 16 + hi * 8;
#pragma unroll
        for (int st = 0; st < 4; ++st) {
            const int J = j0 + st * 16 + m;
#pragma unroll
            for (int r = 0; r < 8; ++r) {
                const float s = acc[rt][st][r];
                const int   I = Ibase + r;
                if (J > I && s > MARGINF) {
                    lsum += s - MARGINF;
                    lcnt += 1u;
                }
            }
        }
    }

    // wave32 reduction
    for (int off = 16; off > 0; off >>= 1) {
        lsum += __shfl_down(lsum, off, 32);
        lcnt += __shfl_down(lcnt, off, 32);
    }
    if (lane == 0 && lcnt != 0u) {
        atomicAdd(accf, lsum);
        atomicAdd(accu, lcnt);
    }
}

// ---------------------------------------------------------------------------
// Kernel 4: finalize scalar.
// ---------------------------------------------------------------------------
__global__ void pd_final(const float* __restrict__ accf,
                         const unsigned int* __restrict__ accu,
                         float* __restrict__ out) {
    const unsigned int cnt = *accu;
    out[0] = (cnt == 0u) ? 0.0f : (*accf) / (float)cnt;
}

// ---------------------------------------------------------------------------
extern "C" void kernel_launch(void* const* d_in, const int* in_sizes, int n_in,
                              void* d_out, int out_size, void* d_ws, size_t ws_size,
                              hipStream_t stream) {
    (void)in_sizes; (void)n_in; (void)out_size; (void)ws_size;

    const float* prompts = (const float*)d_in[0];

    // Workspace layout:
    //   [0, 4096*2048*2)  : f16 normalized matrix, row-major
    //   then float sum accumulator, uint count accumulator
    _Float16* Pf16 = (_Float16*)d_ws;
    char*     wsb  = (char*)d_ws;
    float*        accf = (float*)(wsb + (size_t)NROWS * KDIM * sizeof(_Float16));
    unsigned int* accu = (unsigned int*)(accf + 1);

    pd_normalize<<<NROWS, 256, 0, stream>>>(prompts, Pf16);
    pd_init<<<1, 1, 0, stream>>>(accf, accu);

    const int wavesPerBlock = 8;
    const int blocks = (NBLOCKS + wavesPerBlock - 1) / wavesPerBlock;
    pd_blocks<<<blocks, 32 * wavesPerBlock, 0, stream>>>(Pf16, accf, accu);

    pd_final<<<1, 1, 0, stream>>>(accf, accu, (float*)d_out);
}
